// CausalConv3dFP8_62672162783982
// MI455X (gfx1250) — 8-bit matrix operand order repaired and checked by reading, compiled, not run on hardware
//
#include <hip/hip_runtime.h>


typedef int   v4i  __attribute__((ext_vector_type(4)));
typedef int   v16i __attribute__((ext_vector_type(16)));
typedef float v4f  __attribute__((ext_vector_type(4)));
typedef float v8f  __attribute__((ext_vector_type(8)));

#define BATCH 2
#define CIN   128
#define COUT  128
#define DD    4
#define HH    192
#define WW    320
#define DP    (DD + 2)
#define HHP   (HH + 2)
#define WWP   (WW + 2)

#define XQ_BYTES ((size_t)BATCH * DP * HHP * WWP * CIN)
#define WQ_BYTES (27 * 8 * 32 * 64)

__device__ __forceinline__ unsigned f32_to_e4m3(float f) {
  unsigned u  = __float_as_uint(f);
  unsigned s  = (u >> 24) & 0x80u;
  unsigned au = u & 0x7FFFFFFFu;
  if (au >= 0x7F800000u) return s | 0x7Fu;
  if (au >= 0x43E00000u) return s | 0x7Eu;
  if (au <  0x3A800000u) return s;
  int exp = (int)(au >> 23) - 127;
  unsigned mant = (au & 0x7FFFFFu) | 0x800000u;
  unsigned code;
  if (exp < -6) {
    int shift = 20 + (-6 - exp);
    unsigned keep  = mant >> shift;
    unsigned rem   = mant & ((1u << shift) - 1u);
    unsigned halfv = 1u << (shift - 1);
    keep += (rem > halfv) || (rem == halfv && (keep & 1u));
    code = keep;
  } else {
    unsigned keep  = mant >> 20;
    unsigned rem   = mant & ((1u << 20) - 1u);
    unsigned halfv = 1u << 19;
    keep += (rem > halfv) || (rem == halfv && (keep & 1u));
    code = ((unsigned)(exp + 7) << 3) + keep - 8u;
  }
  if (code > 0x7Eu) code = 0x7Eu;
  return s | code;
}

__device__ __forceinline__ unsigned pack4_e4m3(float a, float b, float c, float d) {
  return  f32_to_e4m3(a)        | (f32_to_e4m3(b) << 8) |
         (f32_to_e4m3(c) << 16) | (f32_to_e4m3(d) << 24);
}

__global__ __launch_bounds__(256) void quant_pack_x(
    const float* __restrict__ x, unsigned char* __restrict__ xq) {
  int row = blockIdx.x;
  const int ph = row % HHP; row /= HHP;
  const int pd = row % DP;  const int b = row / DP;
  const bool zrow = (pd < 2) | (ph == 0) | (ph == HHP - 1);
  const int d = pd - 2, hgt = ph - 1;

  unsigned* dst = (unsigned*)(xq + (size_t)blockIdx.x * (size_t)(WWP * CIN));
  const int cs = DD * HH * WW;

  for (int i = threadIdx.x; i < WWP * (CIN / 4); i += 256) {
    const int w  = i >> 5;
    const int s4 = i & 31;
    unsigned val = 0u;
    if (!zrow && w != 0 && w != WWP - 1) {
      const int q  = s4 << 2;
      const int g  = q >> 5, r = q & 31;
      const int c0 = ((g >> 1) << 6) + ((g & 1) << 3) + ((r >> 3) << 4) + (r & 7);
      const int xi = (((b * CIN + c0) * DD + d) * HH + hgt) * WW + (w - 1);
      const float f0 = __builtin_nontemporal_load(x + xi);
      const float f1 = __builtin_nontemporal_load(x + xi + cs);
      const float f2 = __builtin_nontemporal_load(x + xi + 2 * cs);
      const float f3 = __builtin_nontemporal_load(x + xi + 3 * cs);
      val = pack4_e4m3(f0, f1, f2, f3);
    }
    dst[i] = val;
  }
}

__global__ __launch_bounds__(256) void quant_pack_w(
    const float* __restrict__ wsrc, unsigned char* __restrict__ wq) {
  const int idx  = blockIdx.x * 256 + threadIdx.x;
  const int tap  = idx >> 14;
  const int rem  = idx & 16383;
  const int nt   = rem >> 11;
  const int lane = (rem >> 6) & 31;
  const int r    = rem & 63;
  const int K    = ((r >> 5) << 6) + (((r >> 3) & 3) << 4) + ((lane >> 4) << 3) + (r & 7);
  const int O    = (nt << 4) + (lane & 15);
  const int kd = tap / 9, kh = (tap / 3) % 3, kw = tap % 3;
  const int widx = (((O * CIN + K) * 3 + kd) * 3 + kh) * 3 + kw;
  wq[idx] = (unsigned char)f32_to_e4m3(wsrc[widx]);
}

__global__ __launch_bounds__(256) void conv_wmma_fp8(
    const unsigned char* __restrict__ xq,
    const unsigned char* __restrict__ wq,
    float* __restrict__ out) {
  const int lane = threadIdx.x & 31;
  const int wv   = threadIdx.x >> 5;
  const int wm   = wv & 1;
  const int wn   = wv >> 1;
  const int mrow = lane & 15;
  const int half = lane >> 4;

  int t = blockIdx.x;
  const int wt = t % (WW / 64); t /= (WW / 64);
  const int h  = t % HH;        t /= HH;
  const int d  = t % DD;        const int b = t / DD;
  const int w0 = wt * 64 + wm * 32;

  const int aoff = mrow * CIN + half * 32;
  const unsigned char* bb = wq + ((wn * 2) * 32 + lane) * 64;

  v8f acc00 = {}, acc01 = {}, acc10 = {}, acc11 = {};

#pragma unroll 1
  for (int kd = 0; kd < 3; ++kd) {
    if (kd < 2) {
      const unsigned char* pf =
          xq + ((((b * DP + d + kd + 1) * HHP + h) * WWP) + w0) * CIN + lane * 128;
      __builtin_prefetch(pf, 0, 1);
      __builtin_prefetch(pf + (size_t)(WWP * CIN), 0, 1);
      __builtin_prefetch(pf + (size_t)(2 * WWP * CIN), 0, 1);
    }
#pragma unroll
    for (int kh = 0; kh < 3; ++kh) {
      const int rowb = (((b * DP + d + kd) * HHP + (h + kh)) * WWP) * CIN;
#pragma unroll
      for (int kw = 0; kw < 3; ++kw) {
        const int tap = (kd * 3 + kh) * 3 + kw;
        const unsigned char* ap = xq + rowb + (w0 + kw) * CIN + aoff;
        const unsigned char* bp = bb + tap * (8 * 2048);

        union { v16i v; v4i q[4]; } A0, A1, B0, B1;
        A0.q[0] = *(const v4i*)(ap);
        A0.q[1] = *(const v4i*)(ap + 16);
        A0.q[2] = *(const v4i*)(ap + 64);
        A0.q[3] = *(const v4i*)(ap + 80);
        A1.q[0] = *(const v4i*)(ap + 2048);
        A1.q[1] = *(const v4i*)(ap + 2048 + 16);
        A1.q[2] = *(const v4i*)(ap + 2048 + 64);
        A1.q[3] = *(const v4i*)(ap + 2048 + 80);
        B0.q[0] = *(const v4i*)(bp);
        B0.q[1] = *(const v4i*)(bp + 16);
        B0.q[2] = *(const v4i*)(bp + 32);
        B0.q[3] = *(const v4i*)(bp + 48);
        B1.q[0] = *(const v4i*)(bp + 2048);
        B1.q[1] = *(const v4i*)(bp + 2048 + 16);
        B1.q[2] = *(const v4i*)(bp + 2048 + 32);
        B1.q[3] = *(const v4i*)(bp + 2048 + 48);

        acc00 = __builtin_amdgcn_wmma_f32_16x16x128_fp8_fp8(
                    A0.v, B0.v, (short)0, acc00, false, false);
        acc10 = __builtin_amdgcn_wmma_f32_16x16x128_fp8_fp8(
                    A1.v, B0.v, (short)0, acc10, false, false);
        acc01 = __builtin_amdgcn_wmma_f32_16x16x128_fp8_fp8(
                    A0.v, B1.v, (short)0, acc01, false, false);
        acc11 = __builtin_amdgcn_wmma_f32_16x16x128_fp8_fp8(
                    A1.v, B1.v, (short)0, acc11, false, false);
      }
    }
  }

  const int ostride = DD * HH * WW;
  const int sbase   = (d * HH + h) * WW + w0 + half * 8;
  const int o0      = b * COUT + wn * 32 + mrow;

  union { v8f v; v4f q[2]; } C;
  C.v = acc00;
  { float* p = out + o0 * ostride + sbase;
    __builtin_nontemporal_store(C.q[0], (v4f*)(p));
    __builtin_nontemporal_store(C.q[1], (v4f*)(p + 4)); }
  C.v = acc10;
  { float* p = out + o0 * ostride + sbase + 16;
    __builtin_nontemporal_store(C.q[0], (v4f*)(p));
    __builtin_nontemporal_store(C.q[1], (v4f*)(p + 4)); }
  C.v = acc01;
  { float* p = out + (o0 + 16) * ostride + sbase;
    __builtin_nontemporal_store(C.q[0], (v4f*)(p));
    __builtin_nontemporal_store(C.q[1], (v4f*)(p + 4)); }
  C.v = acc11;
  { float* p = out + (o0 + 16) * ostride + sbase + 16;
    __builtin_nontemporal_store(C.q[0], (v4f*)(p));
    __builtin_nontemporal_store(C.q[1], (v4f*)(p + 4)); }
}

extern "C" void kernel_launch(void* const* d_in, const int* in_sizes, int n_in,
                              void* d_out, int out_size, void* d_ws, size_t ws_size,
                              hipStream_t stream) {
  const float* x = (const float*)d_in[0];
  const float* w = (const float*)d_in[1];
  float* out = (float*)d_out;

  unsigned char* xq = (unsigned char*)d_ws;
  unsigned char* wq = xq + XQ_BYTES;

  quant_pack_x<<<BATCH * DP * HHP, 256, 0, stream>>>(x, xq);
  quant_pack_w<<<WQ_BYTES / 256, 256, 0, stream>>>(w, wq);
  conv_wmma_fp8<<<BATCH * DD * HH * (WW / 64), 256, 0, stream>>>(xq, wq, out);
}
